// TransformerBlock_23776938950760
// MI455X (gfx1250) — matrix operand order repaired and checked by reading, not compiled, not run on hardware
//
#include <hip/hip_runtime.h>
#include <hip/hip_bf16.h>

// ---------------------------------------------------------------------------
// Problem constants (match reference)
// ---------------------------------------------------------------------------
#define B_  2
#define S_  2048
#define D_  2048
#define H_  16
#define HD_ 128
#define I_  8192
#define M_  (B_ * S_)           // 4096 tokens

// ---------------------------------------------------------------------------
// Types for CDNA5 WMMA (wave32)
// ---------------------------------------------------------------------------
typedef __attribute__((ext_vector_type(8)))  float   v8f;
typedef __attribute__((ext_vector_type(8)))  __bf16  v8bf;
typedef __attribute__((ext_vector_type(16))) __bf16  v16bf;

union AFrag { v16bf v; v8bf h8[2]; };
union U16x8 { uint4 u; unsigned short s[8]; };
union U16x4 { uint2 u; unsigned short s[4]; };

__device__ __forceinline__ unsigned short f2bf(float f) {
  unsigned int u = __builtin_bit_cast(unsigned int, f);
  u += 0x7fffu + ((u >> 16) & 1u);          // round-to-nearest-even
  return (unsigned short)(u >> 16);
}
__device__ __forceinline__ float bf2f(unsigned short s) {
  unsigned int u = ((unsigned int)s) << 16;
  return __builtin_bit_cast(float, u);
}

__device__ __forceinline__ v8f wmma_bf16(v16bf a, v16bf b, v8f c) {
  // D = A(16x32 bf16) * B(32x16 bf16) + C(16x16 f32)
  return __builtin_amdgcn_wmma_f32_16x16x32_bf16(
      /*neg_a=*/false, a, /*neg_b=*/false, b,
      /*c_mod=*/(short)0, c, /*reuse_a=*/false, /*reuse_b=*/false);
}

// ---------------------------------------------------------------------------
// CDNA5 async global->LDS DMA (ASYNCcnt-tracked), via inline asm since the
// builtins for this family are not probe-confirmed on either toolchain.
// Per-lane: copies 16B from the lane's global address to the lane's LDS addr.
// Generic LDS pointers carry the wave-relative LDS offset in their low 32 bits
// (aperture in the high half), so a truncating cast yields the VDST operand.
// ---------------------------------------------------------------------------
__device__ __forceinline__ void async_load_b128(unsigned lds_off, const void* gptr) {
  asm volatile("global_load_async_to_lds_b128 %0, %1, off"
               :: "v"(lds_off), "v"((unsigned long long)(size_t)gptr)
               : "memory");
}
__device__ __forceinline__ void wait_async() {
  asm volatile("s_wait_asynccnt 0x0" ::: "memory");
}

// ---------------------------------------------------------------------------
// fp32 -> bf16 elementwise (weights)
// ---------------------------------------------------------------------------
__global__ __launch_bounds__(256) void cvt_f32_bf16(
    const float* __restrict__ src, unsigned short* __restrict__ dst, int n4) {
  int i = blockIdx.x * blockDim.x + threadIdx.x;
  if (i >= n4) return;
  const float4 v = ((const float4*)src)[i];
  U16x4 o;
  o.s[0] = f2bf(v.x); o.s[1] = f2bf(v.y);
  o.s[2] = f2bf(v.z); o.s[3] = f2bf(v.w);
  ((uint2*)dst)[i] = o.u;
}

// ---------------------------------------------------------------------------
// LayerNorm (fp32 in) -> bf16 out.  One 256-thread block (8 waves) per row.
// ---------------------------------------------------------------------------
__global__ __launch_bounds__(256) void ln_bf16_kernel(
    const float* __restrict__ x, const float* __restrict__ gam,
    const float* __restrict__ bet, unsigned short* __restrict__ out) {
  const int row = blockIdx.x;
  const int tid = threadIdx.x;
  const float* xr = x + (size_t)row * D_;

  const float4 a = ((const float4*)xr)[tid * 2];
  const float4 b = ((const float4*)xr)[tid * 2 + 1];
  float s = a.x + a.y + a.z + a.w + b.x + b.y + b.z + b.w;
  float q = a.x*a.x + a.y*a.y + a.z*a.z + a.w*a.w
          + b.x*b.x + b.y*b.y + b.z*b.z + b.w*b.w;
  #pragma unroll
  for (int o = 16; o >= 1; o >>= 1) {
    s += __shfl_xor(s, o, 32);
    q += __shfl_xor(q, o, 32);
  }
  __shared__ float ssum[8], ssq[8];
  if ((tid & 31) == 0) { ssum[tid >> 5] = s; ssq[tid >> 5] = q; }
  __syncthreads();
  float ts = 0.f, tq = 0.f;
  #pragma unroll
  for (int w = 0; w < 8; ++w) { ts += ssum[w]; tq += ssq[w]; }
  const float mean = ts * (1.0f / D_);
  const float var  = tq * (1.0f / D_) - mean * mean;
  const float inv  = rsqrtf(var + 1e-5f);

  const int c0 = tid * 8;
  const float vals[8] = {a.x, a.y, a.z, a.w, b.x, b.y, b.z, b.w};
  U16x8 o8;
  #pragma unroll
  for (int i = 0; i < 8; ++i) {
    float o = (vals[i] - mean) * inv * gam[c0 + i] + bet[c0 + i];
    o8.s[i] = f2bf(o);
  }
  ((uint4*)(out + (size_t)row * D_))[tid] = o8.u;
}

// ---------------------------------------------------------------------------
// WMMA GEMM: C[M,N] = A[M,K] @ W[N,K]^T (+ bias / epilogue)
// A, W bf16 row-major (K contiguous).  128x128x32 tiles, 8 waves/block,
// each wave -> 64x32 via 4x2 tiles of v_wmma_f32_16x16x32_bf16.
// Tiles are DMA'd straight into LDS with global_load_async_to_lds_b128:
// no staging VGPRs, one s_wait_asynccnt + one barrier per K-step.
// ---------------------------------------------------------------------------
enum { MODE_BF16 = 0, MODE_SILU = 1, MODE_MUL = 2, MODE_RES = 3 };

template <int MODE>
__global__ __launch_bounds__(256) void gemm_bf16_kernel(
    const unsigned short* __restrict__ A,
    const unsigned short* __restrict__ Bw,
    const float* __restrict__ bias,
    const unsigned short* __restrict__ aux,   // MODE_MUL: [M,N] bf16
    const float* __restrict__ res,            // MODE_RES: [M,N] f32
    unsigned short* __restrict__ outb,
    float* __restrict__ outf,
    const int Ndim, const int Kdim) {
  constexpr int LD = 48;                      // padded halfs per LDS row
  __shared__ __align__(32) unsigned short sA[2][128 * LD];
  __shared__ __align__(32) unsigned short sB[2][128 * LD];

  const int tid  = threadIdx.x;
  const int lane = tid & 31;
  const int wid  = tid >> 5;
  const int bm = blockIdx.y * 128, bn = blockIdx.x * 128;
  const int wm = (wid & 1) * 64,  wn = (wid >> 1) * 32;
  const int nK = Kdim >> 5;

  // cooperative tile DMA: 512 x 16B chunks per matrix, 2 per thread
  const int c0 = tid * 2, c1 = tid * 2 + 1;
  const int ar0 = c0 >> 2, ak0 = (c0 & 3) * 8;
  const int ar1 = c1 >> 2, ak1 = (c1 & 3) * 8;
  const unsigned short* Abase = A  + (size_t)bm * Kdim;
  const unsigned short* Bbase = Bw + (size_t)bn * Kdim;

  auto issueAsync = [&](int kk, int buf) {
    const int kof = kk * 32;
    async_load_b128((unsigned)(size_t)&sA[buf][ar0 * LD + ak0],
                    Abase + (size_t)ar0 * Kdim + kof + ak0);
    async_load_b128((unsigned)(size_t)&sA[buf][ar1 * LD + ak1],
                    Abase + (size_t)ar1 * Kdim + kof + ak1);
    async_load_b128((unsigned)(size_t)&sB[buf][ar0 * LD + ak0],
                    Bbase + (size_t)ar0 * Kdim + kof + ak0);
    async_load_b128((unsigned)(size_t)&sB[buf][ar1 * LD + ak1],
                    Bbase + (size_t)ar1 * Kdim + kof + ak1);
  };

  v8f acc[4][2] = {};

  issueAsync(0, 0);
  wait_async();
  __syncthreads();

  const int arow = lane & 15;            // A row / B col within 16-tile
  const int k0a  = (lane >> 4) * 16;     // A K-base per half-wave
  const int k0b  = (lane >> 4) * 16;     // B K-base per half-wave

  for (int kk = 0; kk < nK; ++kk) {
    const int cur = kk & 1;
    if (kk + 1 < nK) {
      issueAsync(kk + 1, cur ^ 1);       // fills the buffer last read at kk-1
      if (kk + 2 < nK) {                 // -> global_prefetch_b8
        __builtin_prefetch(Abase + (size_t)ar0 * Kdim + (kk + 2) * 32 + ak0, 0, 1);
        __builtin_prefetch(Bbase + (size_t)ar0 * Kdim + (kk + 2) * 32 + ak0, 0, 1);
      }
    }
    AFrag af[4];
    #pragma unroll
    for (int mi = 0; mi < 4; ++mi) {
      const unsigned short* p = &sA[cur][(wm + mi * 16 + arow) * LD];
      af[mi].h8[0] = *(const v8bf*)(p + k0a);
      af[mi].h8[1] = *(const v8bf*)(p + k0a + 8);
    }
    v16bf bfr[2];
    #pragma unroll
    for (int ni = 0; ni < 2; ++ni)
      bfr[ni] = *(const v16bf*)&sB[cur][(wn + ni * 16 + arow) * LD + k0b];
    #pragma unroll
    for (int mi = 0; mi < 4; ++mi)
      #pragma unroll
      for (int ni = 0; ni < 2; ++ni)
        acc[mi][ni] = wmma_bf16(af[mi].v, bfr[ni], acc[mi][ni]);
    if (kk + 1 < nK) {
      wait_async();                      // my DMA done; barrier makes all visible
      __syncthreads();
    }
  }

  // epilogue.  C layout: VGPR j -> row j (lanes 0-15) / row j+8 (lanes 16-31)
  const int rbase = (lane >> 4) * 8;
  #pragma unroll
  for (int mi = 0; mi < 4; ++mi) {
    #pragma unroll
    for (int ni = 0; ni < 2; ++ni) {
      const int gn = bn + wn + ni * 16 + (lane & 15);
      const float bv = bias[gn];
      #pragma unroll
      for (int j = 0; j < 8; ++j) {
        const int gm = bm + wm + mi * 16 + rbase + j;
        const size_t idx = (size_t)gm * Ndim + gn;
        float c = acc[mi][ni][j] + bv;
        if constexpr (MODE == MODE_SILU) c = c / (1.0f + __expf(-c));
        if constexpr (MODE == MODE_MUL)  c = c * bf2f(aux[idx]);
        if constexpr (MODE == MODE_RES)  outf[idx] = c + res[idx];
        else                             outb[idx] = f2bf(c);
      }
    }
  }
}

// ---------------------------------------------------------------------------
// Flash attention (bf16 in/out, f32 softmax state).
// Block = (b, h, 128 q-rows); 8 waves, each owns 16 q rows.
// Streams 32-key chunks: S = Q K^T (8 WMMA), online softmax, O += P V (8 WMMA)
// K tile is DMA'd into LDS asynchronously; V needs a transpose so it goes
// through VGPRs.
// ---------------------------------------------------------------------------
__global__ __launch_bounds__(256) void attn_kernel(
    const unsigned short* __restrict__ Q, const unsigned short* __restrict__ K,
    const unsigned short* __restrict__ V, unsigned short* __restrict__ O) {
  constexpr float SCALE = 0.08838834764831845f;   // 1/sqrt(128)
  constexpr int LK = 144;   // sK row stride (halfs), 288B (32B multiple)
  constexpr int LV = 48;    // sVt row stride (halfs), 96B (32B multiple)
  __shared__ __align__(32) unsigned short sK[32 * LK];     // [key][d]
  __shared__ __align__(32) unsigned short sVt[128 * LV];   // [d][key]
  __shared__ __align__(32) unsigned short sP[8 * 16 * 32]; // per-wave staging

  const int tid  = threadIdx.x;
  const int lane = tid & 31;
  const int wid  = tid >> 5;
  const int qt = blockIdx.x, h = blockIdx.y, b = blockIdx.z;

  // --- load this wave's 16x128 Q tile as 4 A-fragments (K-chunks of 32) ---
  const int qrow = qt * 128 + wid * 16 + (lane & 15);
  const unsigned short* qptr = Q + ((size_t)(b * S_ + qrow) * H_ + h) * HD_;
  const int koff = (lane >> 4) * 16;
  AFrag qf[4];
  #pragma unroll
  for (int c = 0; c < 4; ++c) {
    qf[c].h8[0] = *(const v8bf*)(qptr + c * 32 + koff);
    qf[c].h8[1] = *(const v8bf*)(qptr + c * 32 + koff + 8);
  }

  v8f oacc[8] = {};
  float rowmax[8], rowsum[8];
  #pragma unroll
  for (int j = 0; j < 8; ++j) { rowmax[j] = -1e30f; rowsum[j] = 0.0f; }

  unsigned short* pst = &sP[wid * 512];
  const int nk  = lane & 15;           // B-frag column index
  const int kdb = (lane >> 4) * 16;    // B-frag K-base per half-wave

  for (int kv0 = 0; kv0 < S_; kv0 += 32) {
    __syncthreads();
    // cooperative load of K (async DMA, row-major) and V (transposed) chunk
    #pragma unroll
    for (int t = 0; t < 2; ++t) {
      const int cid = tid * 2 + t;               // 0..511
      const int key = cid >> 4;                  // 0..31
      const int d0  = (cid & 15) * 8;            // 0..120
      const size_t src = ((size_t)(b * S_ + kv0 + key) * H_ + h) * HD_ + d0;
      async_load_b128((unsigned)(size_t)&sK[key * LK + d0], K + src);
      U16x8 vv; vv.u = *(const uint4*)(V + src);
      #pragma unroll
      for (int i = 0; i < 8; ++i) sVt[(d0 + i) * LV + key] = vv.s[i];
    }
    wait_async();
    __syncthreads();

    // ---- S tile: Q(16x128) @ K_chunk^T(128x32) ----
    v8f s0 = {}, s1 = {};
    #pragma unroll
    for (int c = 0; c < 4; ++c) {
      v16bf bk0 = *(const v16bf*)&sK[nk * LK + c * 32 + kdb];
      v16bf bk1 = *(const v16bf*)&sK[(16 + nk) * LK + c * 32 + kdb];
      s0 = wmma_bf16(qf[c].v, bk0, s0);
      s1 = wmma_bf16(qf[c].v, bk1, s1);
    }

    // ---- online softmax (rows live on 16-lane half-waves) ----
    float fscale[8];
    #pragma unroll
    for (int j = 0; j < 8; ++j) {
      const float a0 = s0[j] * SCALE, a1 = s1[j] * SCALE;
      float cm = fmaxf(a0, a1);
      #pragma unroll
      for (int o = 1; o < 16; o <<= 1) cm = fmaxf(cm, __shfl_xor(cm, o, 16));
      const float mn = fmaxf(rowmax[j], cm);
      const float fs = __expf(rowmax[j] - mn);
      rowmax[j] = mn;
      const float p0 = __expf(a0 - mn), p1 = __expf(a1 - mn);
      float ps = p0 + p1;
      #pragma unroll
      for (int o = 1; o < 16; o <<= 1) ps += __shfl_xor(ps, o, 16);
      rowsum[j] = rowsum[j] * fs + ps;
      fscale[j] = fs;
      const int row = (lane >> 4) * 8 + j;       // C-layout row
      pst[row * 32 + (lane & 15)]      = f2bf(p0);
      pst[row * 32 + 16 + (lane & 15)] = f2bf(p1);
    }
    #pragma unroll
    for (int nt = 0; nt < 8; ++nt)
      #pragma unroll
      for (int j = 0; j < 8; ++j) oacc[nt][j] *= fscale[j];

    // ---- P as A-fragment (same-wave LDS round trip, DS in-order) ----
    AFrag pf;
    {
      const int prow = lane & 15;
      const int pk   = (lane >> 4) * 16;
      pf.h8[0] = *(const v8bf*)&pst[prow * 32 + pk];
      pf.h8[1] = *(const v8bf*)&pst[prow * 32 + pk + 8];
    }

    // ---- O += P(16x32) @ V_chunk(32x128) ----
    #pragma unroll
    for (int nt = 0; nt < 8; ++nt) {
      v16bf bv = *(const v16bf*)&sVt[(nt * 16 + nk) * LV + kdb];
      oacc[nt] = wmma_bf16(pf.v, bv, oacc[nt]);
    }
  }

  // ---- normalize and store [B,S,H,HD] bf16 ----
  #pragma unroll
  for (int nt = 0; nt < 8; ++nt) {
    #pragma unroll
    for (int j = 0; j < 8; ++j) {
      const int row = qt * 128 + wid * 16 + (lane >> 4) * 8 + j;
      const int col = nt * 16 + (lane & 15);
      const float val = oacc[nt][j] / rowsum[j];
      O[((size_t)(b * S_ + row) * H_ + h) * HD_ + col] = f2bf(val);
    }
  }
}

// ---------------------------------------------------------------------------
// Host launch
// ---------------------------------------------------------------------------
extern "C" void kernel_launch(void* const* d_in, const int* in_sizes, int n_in,
                              void* d_out, int out_size, void* d_ws, size_t ws_size,
                              hipStream_t stream) {
  (void)in_sizes; (void)n_in; (void)out_size; (void)ws_size;
  const float* x  = (const float*)d_in[0];
  const float* Wq = (const float*)d_in[1];  const float* bq = (const float*)d_in[2];
  const float* Wk = (const float*)d_in[3];  const float* bk = (const float*)d_in[4];
  const float* Wv = (const float*)d_in[5];  const float* bv = (const float*)d_in[6];
  const float* Wo = (const float*)d_in[7];  const float* bo = (const float*)d_in[8];
  const float* Wg = (const float*)d_in[9];  const float* bg = (const float*)d_in[10];
  const float* Wu = (const float*)d_in[11]; const float* bu = (const float*)d_in[12];
  const float* Wd = (const float*)d_in[13]; const float* bd = (const float*)d_in[14];
  const float* g1 = (const float*)d_in[15]; const float* b1 = (const float*)d_in[16];
  const float* g2 = (const float*)d_in[17]; const float* b2 = (const float*)d_in[18];
  float* out = (float*)d_out;

  char* ws = (char*)d_ws;
  size_t off = 0;
  auto alloc = [&](size_t bytes) {
    char* p = ws + off;
    off += (bytes + 255) & ~(size_t)255;
    return p;
  };
  unsigned short* h1   = (unsigned short*)alloc((size_t)M_ * D_ * 2);
  unsigned short* wqb  = (unsigned short*)alloc((size_t)D_ * D_ * 2);
  unsigned short* wkb  = (unsigned short*)alloc((size_t)D_ * D_ * 2);
  unsigned short* wvb  = (unsigned short*)alloc((size_t)D_ * D_ * 2);
  unsigned short* wob  = (unsigned short*)alloc((size_t)D_ * D_ * 2);
  unsigned short* wgb  = (unsigned short*)alloc((size_t)I_ * D_ * 2);
  unsigned short* wub  = (unsigned short*)alloc((size_t)I_ * D_ * 2);
  unsigned short* wdb  = (unsigned short*)alloc((size_t)D_ * I_ * 2);
  unsigned short* qb   = (unsigned short*)alloc((size_t)M_ * D_ * 2);
  unsigned short* kb   = (unsigned short*)alloc((size_t)M_ * D_ * 2);
  unsigned short* vb   = (unsigned short*)alloc((size_t)M_ * D_ * 2);
  unsigned short* ab   = (unsigned short*)alloc((size_t)M_ * D_ * 2);
  float*          x1   = (float*)         alloc((size_t)M_ * D_ * 4);
  unsigned short* h2   = (unsigned short*)alloc((size_t)M_ * D_ * 2);
  unsigned short* gbuf = (unsigned short*)alloc((size_t)M_ * I_ * 2);
  unsigned short* mbuf = (unsigned short*)alloc((size_t)M_ * I_ * 2);

  const dim3 blk(256);

  // LN1: x -> h1 (bf16)
  ln_bf16_kernel<<<M_, blk, 0, stream>>>(x, g1, b1, h1);

  // weight down-conversion to bf16
  const int n4d = D_ * D_ / 4, n4i = I_ * D_ / 4;
  cvt_f32_bf16<<<n4d / 256, blk, 0, stream>>>(Wq, wqb, n4d);
  cvt_f32_bf16<<<n4d / 256, blk, 0, stream>>>(Wk, wkb, n4d);
  cvt_f32_bf16<<<n4d / 256, blk, 0, stream>>>(Wv, wvb, n4d);
  cvt_f32_bf16<<<n4d / 256, blk, 0, stream>>>(Wo, wob, n4d);
  cvt_f32_bf16<<<n4i / 256, blk, 0, stream>>>(Wg, wgb, n4i);
  cvt_f32_bf16<<<n4i / 256, blk, 0, stream>>>(Wu, wub, n4i);
  cvt_f32_bf16<<<n4i / 256, blk, 0, stream>>>(Wd, wdb, n4i);

  // Q/K/V projections (bf16 out)
  const dim3 gdd(D_ / 128, M_ / 128);
  gemm_bf16_kernel<MODE_BF16><<<gdd, blk, 0, stream>>>(h1, wqb, bq, nullptr, nullptr, qb, nullptr, D_, D_);
  gemm_bf16_kernel<MODE_BF16><<<gdd, blk, 0, stream>>>(h1, wkb, bk, nullptr, nullptr, kb, nullptr, D_, D_);
  gemm_bf16_kernel<MODE_BF16><<<gdd, blk, 0, stream>>>(h1, wvb, bv, nullptr, nullptr, vb, nullptr, D_, D_);

  // attention
  const dim3 gat(S_ / 128, H_, B_);
  attn_kernel<<<gat, blk, 0, stream>>>(qb, kb, vb, ab);

  // O projection + residual -> x1 (f32)
  gemm_bf16_kernel<MODE_RES><<<gdd, blk, 0, stream>>>(ab, wob, bo, nullptr, x, nullptr, x1, D_, D_);

  // LN2: x1 -> h2 (bf16)
  ln_bf16_kernel<<<M_, blk, 0, stream>>>(x1, g2, b2, h2);

  // SwiGLU MLP
  const dim3 gdi(I_ / 128, M_ / 128);
  gemm_bf16_kernel<MODE_SILU><<<gdi, blk, 0, stream>>>(h2, wgb, bg, nullptr, nullptr, gbuf, nullptr, I_, D_);
  gemm_bf16_kernel<MODE_MUL ><<<gdi, blk, 0, stream>>>(h2, wub, bu, gbuf,   nullptr, mbuf, nullptr, I_, D_);

  // down projection + residual -> out (f32)
  gemm_bf16_kernel<MODE_RES><<<gdd, blk, 0, stream>>>(mbuf, wdb, bd, nullptr, x1, nullptr, out, D_, I_);
}
